// MixedLinear_89979564851799
// MI455X (gfx1250) — matrix operand order repaired and checked by reading, not compiled, not run on hardware
//
#include <hip/hip_runtime.h>

// ---------------------------------------------------------------------------
// MixedLinear on MI455X (gfx1250):  out = x @ W^T
//   M=8192, N=4096, K=4096.  K4=3584 cols are fp4(e2m1)*is(e4m3)*os, last
//   K8=512 cols are fp8(e4m3)*pow2 scale.  All fp4/fp8 grid values are exact
//   in bf16, so we dequantize the weights straight to bf16 (one rounding on
//   the scale product), round x to bf16, and run a bf16 WMMA GEMM with fp32
//   accumulation (compute-bound: 275 GFLOP vs 320 MB of HBM traffic).
// Global->LDS staging uses gfx1250 async-to-LDS copies (ASYNCcnt), no VGPR
// staging and no ds_store in the hot loop.
// Workspace layout (96 MB assumed available in d_ws):
//   [0, 32MB)   : W  bf16 [4096][4096]  (row = OUT/N, col = K) -- K-contiguous
//   [32, 96MB)  : A  bf16 [8192][4096]  (row = M,     col = K) -- K-contiguous
// ---------------------------------------------------------------------------

typedef __bf16 bf16;
typedef __attribute__((ext_vector_type(16))) __bf16 v16bf;
typedef __attribute__((ext_vector_type(8)))  float  v8f;

#define M_DIM 8192
#define N_DIM 4096
#define K_DIM 4096
#define K4    3584
#define K8    512

union Pack8 { bf16 h[8]; uint4 u; };
union Frag  { v16bf v; uint4 u[2]; };

// --------------------------- weight dequant to bf16 -------------------------
__global__ __launch_bounds__(256) void dequant_w_kernel(
    const float* __restrict__ q4, const float* __restrict__ os,
    const float* __restrict__ is, const float* __restrict__ q8,
    const float* __restrict__ s8, bf16* __restrict__ W) {
  const int t  = blockIdx.x * 256 + threadIdx.x;   // 2M threads, 8 elems each
  const int n  = t >> 9;                           // OUT row
  const int kk = (t & 511) << 3;                   // K column start (mult of 8)
  float v[8];
  float sc;
  if (kk < K4) {
    sc = is[n * (K4 / 16) + (kk >> 4)] * os[n * (K4 / 128) + (kk >> 7)];
    const float4* p = (const float4*)(q4 + (size_t)n * K4 + kk);
    float4 a = p[0], b = p[1];
    v[0]=a.x; v[1]=a.y; v[2]=a.z; v[3]=a.w; v[4]=b.x; v[5]=b.y; v[6]=b.z; v[7]=b.w;
  } else {
    const int k8 = kk - K4;
    sc = s8[n * (K8 / 32) + (k8 >> 5)];
    const float4* p = (const float4*)(q8 + (size_t)n * K8 + k8);
    float4 a = p[0], b = p[1];
    v[0]=a.x; v[1]=a.y; v[2]=a.z; v[3]=a.w; v[4]=b.x; v[5]=b.y; v[6]=b.z; v[7]=b.w;
  }
  Pack8 pk;
#pragma unroll
  for (int i = 0; i < 8; ++i) pk.h[i] = (bf16)(v[i] * sc);
  *(uint4*)(W + (size_t)n * K_DIM + kk) = pk.u;
}

// --------------------------- activation cast to bf16 ------------------------
__global__ __launch_bounds__(256) void cvt_x_kernel(
    const float* __restrict__ x, bf16* __restrict__ A) {
  const size_t t    = (size_t)blockIdx.x * 256 + threadIdx.x;  // 4M threads
  const size_t base = t * 8;
  const float4* p = (const float4*)(x + base);
  float4 a = p[0], b = p[1];
  Pack8 pk;
  pk.h[0]=(bf16)a.x; pk.h[1]=(bf16)a.y; pk.h[2]=(bf16)a.z; pk.h[3]=(bf16)a.w;
  pk.h[4]=(bf16)b.x; pk.h[5]=(bf16)b.y; pk.h[6]=(bf16)b.z; pk.h[7]=(bf16)b.w;
  *(uint4*)(A + base) = pk.u;
}

// ---------------- gfx1250 async global->LDS copy (ASYNCcnt) -----------------
// VDST = VGPR holding 32-bit LDS byte address (flat-LDS addr[31:0]);
// VADDR = VGPR pair holding 64-bit global address; no SADDR ("off").
__device__ __forceinline__ void async_copy_b128(unsigned lds_addr,
                                                const bf16* gptr) {
  asm volatile("global_load_async_to_lds_b128 %0, %1, off"
               :: "v"(lds_addr), "v"(gptr)
               : "memory");
}
__device__ __forceinline__ void wait_asynccnt0() {
  asm volatile("s_wait_asynccnt 0x0" ::: "memory");
}

// ------------------------------- bf16 WMMA GEMM -----------------------------
// Block: 256 threads (8 waves of 32).  C tile 128x128 per block.
// Wave grid 4(M) x 2(N): each wave owns 32x64 = 2x4 tiles of 16x16.
// LDS: double-buffered A[128][32] and B[128][32] bf16, row stride padded to 40.
#define KSTEP  32
#define LSTR   40        // padded K stride in bf16 elems (80B, breaks conflicts)
#define NSTEPS (K_DIM / KSTEP)   // 128
#define BUFBYTES (128 * LSTR * 2)   // 10240 B per buffer

__global__ __launch_bounds__(256) void gemm_bf16_kernel(
    const bf16* __restrict__ A, const bf16* __restrict__ B,
    float* __restrict__ C) {
  __shared__ __align__(16) bf16 lsA[2][128][LSTR];
  __shared__ __align__(16) bf16 lsB[2][128][LSTR];

  const int tid   = threadIdx.x;
  const int mBase = blockIdx.y * 128;
  const int nBase = blockIdx.x * 128;
  const int w     = tid >> 5;
  const int lane  = tid & 31;
  const int half  = lane >> 4;   // 0: lanes 0-15, 1: lanes 16-31
  const int l16   = lane & 15;
  const int mw    = (w >> 1) * 32;
  const int nw    = (w & 1) * 64;

  // Loader mapping: 128 rows x 32 bf16 = 512 chunks of 16B; 2 chunks/thread.
  const int c0 = tid, c1 = tid + 256;
  const int r0 = c0 >> 2, o0 = (c0 & 3) << 3;
  const int r1 = c1 >> 2, o1 = (c1 & 3) << 3;
  const bf16* aG0 = A + (size_t)(mBase + r0) * K_DIM + o0;
  const bf16* aG1 = A + (size_t)(mBase + r1) * K_DIM + o1;
  const bf16* bG0 = B + (size_t)(nBase + r0) * K_DIM + o0;
  const bf16* bG1 = B + (size_t)(nBase + r1) * K_DIM + o1;

  // LDS byte addresses of this thread's 4 destination chunks in buffer 0.
  const unsigned lA0 = (unsigned)(uintptr_t)&lsA[0][r0][o0];
  const unsigned lA1 = (unsigned)(uintptr_t)&lsA[0][r1][o1];
  const unsigned lB0 = (unsigned)(uintptr_t)&lsB[0][r0][o0];
  const unsigned lB1 = (unsigned)(uintptr_t)&lsB[0][r1][o1];

  v8f acc[2][4];
#pragma unroll
  for (int mi = 0; mi < 2; ++mi)
#pragma unroll
    for (int ni = 0; ni < 4; ++ni)
      acc[mi][ni] = (v8f){0.f, 0.f, 0.f, 0.f, 0.f, 0.f, 0.f, 0.f};

  // Prologue: async-fill buffer 0 with K-tile 0.
  async_copy_b128(lA0, aG0);
  async_copy_b128(lA1, aG1);
  async_copy_b128(lB0, bG0);
  async_copy_b128(lB1, bG1);
  wait_asynccnt0();
  __syncthreads();

  for (int kt = 0; kt < NSTEPS; ++kt) {
    const int cur = kt & 1;
    if (kt + 1 < NSTEPS) {       // async-fill the other buffer with tile kt+1
      const unsigned bo = (unsigned)((kt + 1) & 1) * BUFBYTES;
      const size_t   ko = (size_t)(kt + 1) * KSTEP;
      async_copy_b128(lA0 + bo, aG0 + ko);
      async_copy_b128(lA1 + bo, aG1 + ko);
      async_copy_b128(lB0 + bo, bG0 + ko);
      async_copy_b128(lB1 + bo, bG1 + ko);
      if (kt + 3 < NSTEPS) {     // L2 prefetch two tiles ahead
        __builtin_prefetch(aG0 + (size_t)(kt + 3) * KSTEP, 0, 1);
        __builtin_prefetch(bG0 + (size_t)(kt + 3) * KSTEP, 0, 1);
      }
    }

    // A fragments: lane<16 -> M=l16, K {0..7,16..23}; lane>=16 -> K {8..15,24..31}
    v16bf af[2];
#pragma unroll
    for (int mi = 0; mi < 2; ++mi) {
      const bf16* rp = &lsA[cur][mw + mi * 16 + l16][0];
      Frag f;
      f.u[0] = *(const uint4*)(rp + half * 8);
      f.u[1] = *(const uint4*)(rp + 16 + half * 8);
      af[mi] = f.v;
    }
    // B fragments: N=l16, lane<16 -> K {0..7,16..23}; lane>=16 -> K {8..15,24..31}
    v16bf bfr[4];
#pragma unroll
    for (int ni = 0; ni < 4; ++ni) {
      const bf16* rp = &lsB[cur][nw + ni * 16 + l16][0];
      Frag f;
      f.u[0] = *(const uint4*)(rp + half * 8);
      f.u[1] = *(const uint4*)(rp + 16 + half * 8);
      bfr[ni] = f.v;
    }
#pragma unroll
    for (int mi = 0; mi < 2; ++mi)
#pragma unroll
      for (int ni = 0; ni < 4; ++ni)
        acc[mi][ni] = __builtin_amdgcn_wmma_f32_16x16x32_bf16(
            false, af[mi], false, bfr[ni], (short)0, acc[mi][ni], false, false);

    // Wave waits for its own async copies; barrier makes them group-visible.
    if (kt + 1 < NSTEPS) wait_asynccnt0();
    __syncthreads();
  }

  // Epilogue: C/D layout = VGPR i: lanes0-15 -> (M=i, N=lane), lanes16-31 -> (M=8+i)
#pragma unroll
  for (int mi = 0; mi < 2; ++mi) {
#pragma unroll
    for (int i = 0; i < 8; ++i) {
      const int row = mBase + mw + mi * 16 + half * 8 + i;
      float* cp = C + (size_t)row * N_DIM + (nBase + nw + l16);
      cp[0]  = acc[mi][0][i];
      cp[16] = acc[mi][1][i];
      cp[32] = acc[mi][2][i];
      cp[48] = acc[mi][3][i];
    }
  }
}

// ------------------------------------ host ----------------------------------
extern "C" void kernel_launch(void* const* d_in, const int* in_sizes, int n_in,
                              void* d_out, int out_size, void* d_ws, size_t ws_size,
                              hipStream_t stream) {
  const float* x   = (const float*)d_in[0];   // [8192,4096]
  const float* wq4 = (const float*)d_in[1];   // [4096,3584]
  const float* wos = (const float*)d_in[2];   // [4096,28]
  const float* wis = (const float*)d_in[3];   // [4096,224]
  // d_in[4] = w_t (pre-dequantized fp32) -- unused; we dequantize from the
  // quantized tensors, which are already [OUT,K]-major (no transpose needed).
  const float* wq8 = (const float*)d_in[5];   // [4096,512]
  const float* ws8 = (const float*)d_in[6];   // [4096,16]
  float* out = (float*)d_out;

  bf16* Wbf = (bf16*)d_ws;                                        // 32 MB
  bf16* Abf = (bf16*)((char*)d_ws + (size_t)N_DIM * K_DIM * 2);   // 64 MB

  dequant_w_kernel<<<(N_DIM * (K_DIM / 8)) / 256, 256, 0, stream>>>(
      wq4, wos, wis, wq8, ws8, Wbf);
  cvt_x_kernel<<<((size_t)M_DIM * (K_DIM / 8)) / 256, 256, 0, stream>>>(x, Abf);

  dim3 grid(N_DIM / 128, M_DIM / 128);   // (32, 64)
  gemm_bf16_kernel<<<grid, 256, 0, stream>>>(Abf, Wbf, out);
}
